// GAT_56487409877013
// MI455X (gfx1250) — matrix operand order repaired and checked by reading, not compiled, not run on hardware
//
#include <hip/hip_runtime.h>
#include <hip/hip_bf16.h>
#include <stdint.h>

// ---------------------------------------------------------------------------
// Dense GAT (2 layers) for gfx1250 / MI455X, wave32 + v_wmma_f32_16x16x32_f16.
// N=4096 nodes, F_IN=256, layer1: 8 heads x 64, layer2: 1 head x 128.
// ---------------------------------------------------------------------------

typedef _Float16 half_t;
typedef __attribute__((ext_vector_type(16))) _Float16 v16h;
typedef __attribute__((ext_vector_type(8)))  _Float16 v8h;
typedef __attribute__((ext_vector_type(8)))  float    v8f;

#define NODES 4096
#define FIN   256
#define HID   64
#define H1    8
#define F1    512     // H1*HID
#define FOUT  128

// ---------------- WMMA fragment loaders (16-bit, 16x16x32) -----------------
// A 16x32 (MxK): lane L<16 -> m=L, k = 0..15; lanes 16..31 -> k = 16..31.
// Element e of v16h maps to k(e) = e + 16*halfg  -> two
// contiguous 8-half runs.
__device__ __forceinline__ v16h load_a_frag(const half_t* __restrict__ rowk,
                                            int halfg) {
  v8h lo = *(const v8h*)(rowk + 16 * halfg);
  v8h hi = *(const v8h*)(rowk + 16 * halfg + 8);
  v16h r;
#pragma unroll
  for (int e = 0; e < 8; ++e) { r[e] = lo[e]; r[e + 8] = hi[e]; }
  return r;
}
// B 32x16 (KxN), row-major-transposed source BT[n][k]: lane L<16 -> n=L,
// k = 0..15; lanes 16..31 -> k = 16..31.  One contiguous 16-half load.
__device__ __forceinline__ v16h load_b_frag(const half_t* __restrict__ rowk,
                                            int halfg) {
  return *(const v16h*)(rowk + 16 * halfg);
}

// ---------------------- elementwise converts -------------------------------
__global__ void cast_f16_kernel(const float* __restrict__ src,
                                half_t* __restrict__ dst, int n) {
  int i = blockIdx.x * blockDim.x + threadIdx.x;
  if (i < n) dst[i] = (half_t)src[i];
}

// src [K][NC] (row-major) -> dst [NC][K] f16
__global__ void transpose_f16_kernel(const float* __restrict__ src,
                                     half_t* __restrict__ dst, int K, int NC) {
  int i = blockIdx.x * blockDim.x + threadIdx.x;
  if (i >= K * NC) return;
  int n = i / K, k = i - n * K;
  dst[i] = (half_t)src[(size_t)k * NC + n];
}

// --------------- GEMM: C = A(f16,[M][K]) * B (BT f16,[NC][K]) --------------
// Each wave computes a 16(M) x 64(N) strip; output written TRANSPOSED:
// CT [NC][M] f16  (channel-major, so later kernels get contiguous j/k reads).
__global__ __launch_bounds__(256) void gemm_ht_wmma(
    const half_t* __restrict__ A, const half_t* __restrict__ BT,
    half_t* __restrict__ CT, int M, int K, int NC) {
  int wid   = (blockIdx.x * blockDim.x + threadIdx.x) >> 5;
  int lane  = threadIdx.x & 31;
  int halfg = lane >> 4;
  int ln    = lane & 15;
  int nStrips = NC >> 6;
  int mtile = wid / nStrips;
  int ns    = wid - mtile * nStrips;
  int mbase = mtile * 16;
  int nbase = ns * 64;
  if (mbase >= M) return;

  v8f acc0, acc1, acc2, acc3;
#pragma unroll
  for (int r = 0; r < 8; ++r) { acc0[r] = 0.f; acc1[r] = 0.f; acc2[r] = 0.f; acc3[r] = 0.f; }

  const half_t* Arow = A + (size_t)(mbase + ln) * K;
  const half_t* Brow = BT + (size_t)(nbase + ln) * K;
  for (int kb = 0; kb < K; kb += 32) {
    v16h af = load_a_frag(Arow + kb, halfg);
    v16h b0 = load_b_frag(Brow + kb, halfg);
    v16h b1 = load_b_frag(Brow + (size_t)16 * K + kb, halfg);
    v16h b2 = load_b_frag(Brow + (size_t)32 * K + kb, halfg);
    v16h b3 = load_b_frag(Brow + (size_t)48 * K + kb, halfg);
    acc0 = __builtin_amdgcn_wmma_f32_16x16x32_f16(false, af, false, b0, (short)0, acc0, false, false);
    acc1 = __builtin_amdgcn_wmma_f32_16x16x32_f16(false, af, false, b1, (short)0, acc1, false, false);
    acc2 = __builtin_amdgcn_wmma_f32_16x16x32_f16(false, af, false, b2, (short)0, acc2, false, false);
    acc3 = __builtin_amdgcn_wmma_f32_16x16x32_f16(false, af, false, b3, (short)0, acc3, false, false);
  }
  // C element (vgpr r, lane): row m = mbase + r + 8*halfg, col n = nbase+t*16+ln.
  // Store transposed: CT[n][m]; the 8 r-values are contiguous -> one v8h store.
#pragma unroll
  for (int t = 0; t < 4; ++t) {
    v8f a = (t == 0) ? acc0 : (t == 1) ? acc1 : (t == 2) ? acc2 : acc3;
    v8h o;
#pragma unroll
    for (int r = 0; r < 8; ++r) o[r] = (half_t)a[r];
    *(v8h*)(CT + (size_t)(nbase + t * 16 + ln) * M + mbase + 8 * halfg) = o;
  }
}

// --------- per-node attention dots: a = <h[n,head,:], att[head,:]> ---------
// hT is channel-major [H*CH][N]; outputs head-major [H][N] (coalesced).
template <int CH>
__global__ void attn_dots_kernel(const half_t* __restrict__ hT,
                                 const float* __restrict__ attS,
                                 const float* __restrict__ attD,
                                 float* __restrict__ asrcT,
                                 float* __restrict__ adstT, int N) {
  int n = blockIdx.x * blockDim.x + threadIdx.x;
  int h = blockIdx.y;
  if (n >= N) return;
  float s = 0.f, d = 0.f;
  const half_t* base = hT + (size_t)h * CH * N + n;
#pragma unroll 8
  for (int c = 0; c < CH; ++c) {
    float hv = (float)base[(size_t)c * N];
    s += hv * attS[h * CH + c];
    d += hv * attD[h * CH + c];
  }
  asrcT[(size_t)h * N + n] = s;
  adstT[(size_t)h * N + n] = d;
}

// ------------- softmax stats pass: row-max m_i and 1/sum(exp) --------------
// One wave per row i, lanes stride over neighbors j, H heads kept in regs.
template <int H>
__global__ __launch_bounds__(256) void gat_stats_kernel(
    const float* __restrict__ adj, const float* __restrict__ asrcT,
    const float* __restrict__ adstT, float* __restrict__ mrows,
    float* __restrict__ linvs, int N) {
  int i = (blockIdx.x * blockDim.x + threadIdx.x) >> 5;
  int lane = threadIdx.x & 31;
  if (i >= N) return;
  float mm[H], ll[H], ad[H];
#pragma unroll
  for (int h = 0; h < H; ++h) {
    mm[h] = -1e30f; ll[h] = 0.f;
    ad[h] = adstT[(size_t)h * N + i];
  }
  const float* arow = adj + (size_t)i * N;
  for (int j = lane; j < N; j += 32) {
    float a = arow[j];
    if (a != 0.f) {
#pragma unroll
      for (int h = 0; h < H; ++h) {
        float s = asrcT[(size_t)h * N + j] + ad[h];
        s = (s > 0.f) ? s : 0.2f * s;          // leaky_relu(0.2)
        if (s > mm[h]) { ll[h] *= __expf(mm[h] - s); mm[h] = s; }
        ll[h] += __expf(s - mm[h]);
      }
    }
  }
#pragma unroll
  for (int off = 16; off >= 1; off >>= 1) {
#pragma unroll
    for (int h = 0; h < H; ++h) {
      float om = __shfl_xor(mm[h], off, 32);
      float ol = __shfl_xor(ll[h], off, 32);
      float nm = fmaxf(mm[h], om);
      float la = (ll[h] > 0.f) ? ll[h] * __expf(mm[h] - nm) : 0.f;
      float lb = (ol    > 0.f) ? ol    * __expf(om    - nm) : 0.f;
      mm[h] = nm; ll[h] = la + lb;
    }
  }
  if (lane == 0) {
#pragma unroll
    for (int h = 0; h < H; ++h) {
      mrows[(size_t)h * N + i] = mm[h];
      linvs[(size_t)h * N + i] = 1.f / ll[h];
    }
  }
}

// ------------------ fused masked-softmax @ h aggregation -------------------
// Block = 8 waves, 32 output rows (2 row-tiles of 16).  Wave covers CW
// channels (CW/16 c-tiles) of head = cbase/CH.  adj tile + a_src tile are
// staged in LDS once per 32-j step and shared by all waves.  p is built
// directly in the WMMA A-fragment layout; B (h, channel-major) is a single
// contiguous v16h load.  Epilogue: *1/l + bias, leaky_relu(0.01).
template <int CW, int CH, bool HALF_OUT, int CTOT>
__global__ __launch_bounds__(256) void gat_agg_kernel(
    const float* __restrict__ adj, const float* __restrict__ asrcT,
    const float* __restrict__ adstT, const float* __restrict__ mrows,
    const float* __restrict__ linvs, const half_t* __restrict__ hT,
    const float* __restrict__ bias, void* __restrict__ outp, int N) {
  constexpr int NH = (8 * CW) / CH;   // heads spanned by this block
  constexpr int NT = CW / 16;         // c-tiles per wave
  __shared__ float adjT[32][33];
  __shared__ float srcT[NH][32];

  int tid   = threadIdx.x;
  int wave  = tid >> 5;
  int lane  = tid & 31;
  int halfg = lane >> 4;
  int ln    = lane & 15;
  int rowBase = blockIdx.x * 32;
  int cbase   = wave * CW;
  int head    = cbase / CH;

  float adst0 = adstT[(size_t)head * N + rowBase + ln];
  float adst1 = adstT[(size_t)head * N + rowBase + 16 + ln];
  float mI0   = mrows[(size_t)head * N + rowBase + ln];
  float mI1   = mrows[(size_t)head * N + rowBase + 16 + ln];

  v8f acc[2][NT];
#pragma unroll
  for (int mt = 0; mt < 2; ++mt)
#pragma unroll
    for (int t = 0; t < NT; ++t)
#pragma unroll
      for (int r = 0; r < 8; ++r) acc[mt][t][r] = 0.f;

  for (int jb = 0; jb < N; jb += 32) {
    __syncthreads();
    for (int idx = tid; idx < 32 * 32; idx += 256) {
      int r = idx >> 5, c = idx & 31;
      adjT[r][c] = adj[(size_t)(rowBase + r) * N + jb + c];
    }
    for (int idx = tid; idx < NH * 32; idx += 256) {
      int hh = idx >> 5, c = idx & 31;
      srcT[hh][c] = asrcT[(size_t)hh * N + jb + c];
    }
    __syncthreads();

    v16h pa[2];
#pragma unroll
    for (int mt = 0; mt < 2; ++mt) {
      float adsti = mt ? adst1 : adst0;
      float mi    = mt ? mI1 : mI0;
      int grow    = mt * 16 + ln;
#pragma unroll
      for (int e = 0; e < 16; ++e) {
        int k = e + 16 * halfg;                      // A-fragment k pattern
        float sc = srcT[head][k] + adsti;
        sc = (sc > 0.f) ? sc : 0.2f * sc;            // leaky_relu(0.2)
        float pv = (adjT[grow][k] != 0.f) ? __expf(sc - mi) : 0.f;
        pa[mt][e] = (half_t)pv;
      }
    }
#pragma unroll
    for (int t = 0; t < NT; ++t) {
      const half_t* hrow =
          hT + (size_t)(cbase + t * 16 + ln) * N + jb + 16 * halfg;
      v16h bf = *(const v16h*)hrow;
      acc[0][t] = __builtin_amdgcn_wmma_f32_16x16x32_f16(false, pa[0], false, bf, (short)0, acc[0][t], false, false);
      acc[1][t] = __builtin_amdgcn_wmma_f32_16x16x32_f16(false, pa[1], false, bf, (short)0, acc[1][t], false, false);
    }
  }

#pragma unroll
  for (int mt = 0; mt < 2; ++mt) {
    float lv[8];
#pragma unroll
    for (int r = 0; r < 8; ++r)
      lv[r] = linvs[(size_t)head * N + rowBase + mt * 16 + 8 * halfg + r];
#pragma unroll
    for (int t = 0; t < NT; ++t) {
      int c = cbase + t * 16 + ln;
      float bc = bias[c];
#pragma unroll
      for (int r = 0; r < 8; ++r) {
        int i = rowBase + mt * 16 + 8 * halfg + r;
        float v = acc[mt][t][r] * lv[r] + bc;
        v = (v > 0.f) ? v : 0.01f * v;               // leaky_relu(0.01)
        if (HALF_OUT) ((half_t*)outp)[(size_t)i * CTOT + c] = (half_t)v;
        else          ((float*)outp)[(size_t)i * CTOT + c]  = v;
      }
    }
  }
}

// ---------------------------------------------------------------------------
extern "C" void kernel_launch(void* const* d_in, const int* in_sizes, int n_in,
                              void* d_out, int out_size, void* d_ws,
                              size_t ws_size, hipStream_t stream) {
  (void)in_sizes; (void)n_in; (void)out_size; (void)ws_size;
  const int N = NODES;
  const float* x   = (const float*)d_in[0];
  const float* adj = (const float*)d_in[1];
  const float* w1  = (const float*)d_in[2];
  const float* as1 = (const float*)d_in[3];
  const float* ad1 = (const float*)d_in[4];
  const float* b1  = (const float*)d_in[5];
  const float* w2  = (const float*)d_in[6];
  const float* as2 = (const float*)d_in[7];
  const float* ad2 = (const float*)d_in[8];
  const float* b2  = (const float*)d_in[9];
  float* out = (float*)d_out;

  // workspace carve-out (~12.5 MB), 256B aligned chunks
  size_t off = 0;
  char* wsb = (char*)d_ws;
  auto take = [&](size_t bytes) -> void* {
    void* p = wsb + off;
    off += (bytes + 255) & ~(size_t)255;
    return p;
  };
  half_t* xh  = (half_t*)take((size_t)N * FIN * 2);
  half_t* w1t = (half_t*)take((size_t)F1 * FIN * 2);
  half_t* w2t = (half_t*)take((size_t)FOUT * F1 * 2);
  half_t* h1t = (half_t*)take((size_t)F1 * N * 2);    // (x@w1)^T  [512][N]
  half_t* y1h = (half_t*)take((size_t)N * F1 * 2);    // layer1 out [N][512]
  half_t* h2t = (half_t*)take((size_t)FOUT * N * 2);  // (y1@w2)^T [128][N]
  float* a1s = (float*)take((size_t)H1 * N * 4);
  float* a1d = (float*)take((size_t)H1 * N * 4);
  float* m1  = (float*)take((size_t)H1 * N * 4);
  float* l1  = (float*)take((size_t)H1 * N * 4);
  float* a2s = (float*)take((size_t)N * 4);
  float* a2d = (float*)take((size_t)N * 4);
  float* m2  = (float*)take((size_t)N * 4);
  float* l2  = (float*)take((size_t)N * 4);

  // 1) convert inputs to f16 (weights transposed: BT[n][k])
  cast_f16_kernel<<<(N * FIN + 255) / 256, 256, 0, stream>>>(x, xh, N * FIN);
  transpose_f16_kernel<<<(F1 * FIN + 255) / 256, 256, 0, stream>>>(w1, w1t, FIN, F1);
  transpose_f16_kernel<<<(FOUT * F1 + 255) / 256, 256, 0, stream>>>(w2, w2t, F1, FOUT);

  // 2) h1^T = (x @ w1)^T   [512][N]
  {
    int waves = (N / 16) * (F1 / 64);               // 2048
    gemm_ht_wmma<<<waves / 8, 256, 0, stream>>>(xh, w1t, h1t, N, FIN, F1);
  }
  // 3) attention dots + softmax stats (layer 1)
  attn_dots_kernel<HID><<<dim3(N / 256, H1), 256, 0, stream>>>(h1t, as1, ad1, a1s, a1d, N);
  gat_stats_kernel<H1><<<N / 8, 256, 0, stream>>>(adj, a1s, a1d, m1, l1, N);
  // 4) fused softmax @ h1 + bias + leaky_relu(0.01) -> y1 f16 [N][512]
  gat_agg_kernel<64, 64, true, F1><<<N / 32, 256, 0, stream>>>(
      adj, a1s, a1d, m1, l1, h1t, b1, (void*)y1h, N);

  // 5) h2^T = (y1 @ w2)^T  [128][N]
  {
    int waves = (N / 16) * (FOUT / 64);             // 512
    gemm_ht_wmma<<<waves / 8, 256, 0, stream>>>(y1h, w2t, h2t, N, F1, FOUT);
  }
  // 6) layer-2 dots + stats (1 head)
  attn_dots_kernel<FOUT><<<dim3(N / 256, 1), 256, 0, stream>>>(h2t, as2, ad2, a2s, a2d, N);
  gat_stats_kernel<1><<<N / 8, 256, 0, stream>>>(adj, a2s, a2d, m2, l2, N);
  // 7) final aggregation + bias + leaky_relu(0.01) -> d_out f32 [N][128]
  gat_agg_kernel<16, 128, false, FOUT><<<N / 32, 256, 0, stream>>>(
      adj, a2s, a2d, m2, l2, h2t, b2, (void*)out, N);
}
